// CausalAttention_644245094991
// MI455X (gfx1250) — compile-verified, matrix operand order checked by reading, not run on hardware
//
#include <hip/hip_runtime.h>
#include <hip/hip_bf16.h>

typedef __bf16 v16bf __attribute__((ext_vector_type(16)));
typedef float  v8f   __attribute__((ext_vector_type(8)));
typedef unsigned int u32x4 __attribute__((ext_vector_type(4)));
typedef int          i32x8 __attribute__((ext_vector_type(8)));
typedef int          i32x4 __attribute__((ext_vector_type(4)));

#define S_LEN   2048
#define D_DIM   1024
#define NBATCH  4
#define NTOK    (NBATCH * S_LEN)         // 8192 tokens total

#if defined(__has_builtin)
#if __has_builtin(__builtin_amdgcn_tensor_load_to_lds) && __has_builtin(__builtin_amdgcn_s_wait_tensorcnt)
#define USE_TDM 1
#endif
#endif

static __device__ __forceinline__ v8f wmma_bf16(v16bf a, v16bf b, v8f c) {
    return __builtin_amdgcn_wmma_f32_16x16x32_bf16(
        /*neg_a=*/false, a, /*neg_b=*/false, b,
        /*c_mod=*/(short)0, c, /*reuse_a=*/false, /*reuse_b=*/false);
}

// ---------------------------------------------------------------------------
// Kernel 1: x f32 -> bf16
// ---------------------------------------------------------------------------
__global__ void cvt_x_kernel(const float* __restrict__ x, __bf16* __restrict__ xb, int n) {
    int i = blockIdx.x * blockDim.x + threadIdx.x;
    if (i < n) xb[i] = (__bf16)x[i];
}

// ---------------------------------------------------------------------------
// Kernel 2: W[k][n] f32 -> Wt[n][k] bf16 (3 weights concatenated)
// ---------------------------------------------------------------------------
__global__ void cvt_w_kernel(const float* __restrict__ Wq,
                             const float* __restrict__ Wk,
                             const float* __restrict__ Wv,
                             __bf16* __restrict__ wt) {
    int i = blockIdx.x * blockDim.x + threadIdx.x;          // 3 * 1024 * 1024
    if (i >= 3 * D_DIM * D_DIM) return;
    int wsel = i >> 20;
    int rem  = i & ((1 << 20) - 1);
    int n = rem >> 10;
    int k = rem & (D_DIM - 1);
    const float* src = (wsel == 0) ? Wq : (wsel == 1) ? Wk : Wv;
    wt[i] = (__bf16)src[k * D_DIM + n];                     // wt[wsel][n][k]
}

// ---------------------------------------------------------------------------
// Kernel 3: QKV projection GEMM (bf16 WMMA, f32 acc), software-pipelined
//   wsel 0 -> Q row-major, 1 -> K row-major, 2 -> V transposed per batch
// Grid: (256 Mtiles, 4 Nchunks, 3 weights), block 256 (8 waves)
// ---------------------------------------------------------------------------
__global__ void __launch_bounds__(256)
gemm_qkv_kernel(const __bf16* __restrict__ xb, const __bf16* __restrict__ wt,
                __bf16* __restrict__ Qb, __bf16* __restrict__ Kb, __bf16* __restrict__ Vt) {
    const int tid   = threadIdx.x;
    const int w     = tid >> 5;
    const int lane  = tid & 31;
    const int l16   = lane & 15;
    const int lhalf = lane >> 4;

    const int wsel = blockIdx.z;
    const __bf16* __restrict__ W = wt + (size_t)wsel * D_DIM * D_DIM;

    const int m0 = blockIdx.x * 32 + (w >> 2) * 16;
    const int n0 = blockIdx.y * 256 + (w & 3) * 64;

    const __bf16* arow = xb + (size_t)(m0 + l16) * D_DIM + lhalf * 16;
    const __bf16* brow[4];
#pragma unroll
    for (int t = 0; t < 4; ++t)
        brow[t] = W + (size_t)(n0 + t * 16 + l16) * D_DIM + lhalf * 16;

    // prime the pipeline
    v16bf a0 = *(const v16bf*)(arow);
    v16bf b0[4];
#pragma unroll
    for (int t = 0; t < 4; ++t) b0[t] = *(const v16bf*)(brow[t]);

    v8f acc[4] = {};
#pragma unroll 4
    for (int kk = 0; kk < D_DIM; kk += 32) {
        const int kn = (kk + 32) & (D_DIM - 1);   // wraps on last iter (harmless)
        v16bf a1 = *(const v16bf*)(arow + kn);
        v16bf b1[4];
#pragma unroll
        for (int t = 0; t < 4; ++t) b1[t] = *(const v16bf*)(brow[t] + kn);
#pragma unroll
        for (int t = 0; t < 4; ++t) acc[t] = wmma_bf16(a0, b0[t], acc[t]);
        a0 = a1;
#pragma unroll
        for (int t = 0; t < 4; ++t) b0[t] = b1[t];
    }

    if (wsel < 2) {
        __bf16* __restrict__ out = (wsel == 0) ? Qb : Kb;
#pragma unroll
        for (int t = 0; t < 4; ++t)
#pragma unroll
            for (int r = 0; r < 8; ++r) {
                int m = m0 + r + 8 * lhalf;
                out[(size_t)m * D_DIM + n0 + t * 16 + l16] = (__bf16)acc[t][r];
            }
    } else {
        // V transposed: Vt[b][feature][token]
#pragma unroll
        for (int t = 0; t < 4; ++t)
#pragma unroll
            for (int r = 0; r < 8; ++r) {
                int m  = m0 + r + 8 * lhalf;
                int bb = m >> 11;              // m / 2048
                int ml = m & (S_LEN - 1);
                Vt[(size_t)bb * D_DIM * S_LEN + (size_t)(n0 + t * 16 + l16) * S_LEN + ml] =
                    (__bf16)acc[t][r];
            }
    }
}

// ---------------------------------------------------------------------------
// Kernel 4: causal flash attention (TDM Q staging, pipelined WMMA loops)
// Grid: (128 q-tiles, 4 batches), block 256 (8 waves)
// ---------------------------------------------------------------------------
__global__ void __launch_bounds__(256)
flash_attn_kernel(const __bf16* __restrict__ Qb, const __bf16* __restrict__ Kb,
                  const __bf16* __restrict__ Vt, float* __restrict__ out) {
    __shared__ __bf16 sQ[16 * D_DIM];        // 32 KB: query tile
    __shared__ __bf16 sP[16 * 128];          //  4 KB: softmaxed probs tile
    __shared__ float  sm_m[16], sm_l[16], sm_alpha[16];
    __shared__ float  sm_pmax[8 * 16], sm_psum[8 * 16];

    const int tid   = threadIdx.x;
    const int w     = tid >> 5;
    const int lane  = tid & 31;
    const int l16   = lane & 15;
    const int lhalf = lane >> 4;

    const int b  = blockIdx.y;
    const int q0 = blockIdx.x * 16;
    const float scale = 0.03125f;            // 1/sqrt(1024)

    // ---- stage Q tile (16 x 1024 bf16 = 32 KB, contiguous) into LDS ----
#ifdef USE_TDM
    if (w == 0) {
        // Tensor DMA descriptor (D#): 1-D tile of 16384 bf16 elements
        const __bf16* gq = Qb + (size_t)(b * S_LEN + q0) * D_DIM;
        unsigned long long ga  = (unsigned long long)(uintptr_t)gq;
        unsigned int       lds = (unsigned int)(uintptr_t)&sQ[0];
        u32x4 g0;
        g0[0] = 1u;                                          // count=1, user mode
        g0[1] = lds;                                         // lds_addr
        g0[2] = (unsigned int)(ga & 0xffffffffu);            // global_addr[31:0]
        g0[3] = (unsigned int)((ga >> 32) & 0x01ffffffu)     // global_addr[56:32]
              | (2u << 30);                                  // type=2 (image)
        i32x8 g1;
        g1[0] = 0x00010000;                                  // data_size=1 (2 bytes)
        g1[1] = (int)(16384u << 16);                         // tensor_dim0[15:0]=16384
        g1[2] = (int)(1u << 16);                             // tensor_dim1=1
        g1[3] = (int)(16384u << 16);                         // tile_dim0=16384
        g1[4] = 1;                                           // tile_dim1=1
        g1[5] = 16384;                                       // tensor_dim0_stride
        g1[6] = 0;
        g1[7] = 0;
        i32x4 z4 = {0, 0, 0, 0};
        i32x8 z8 = {0, 0, 0, 0, 0, 0, 0, 0};
        __builtin_amdgcn_tensor_load_to_lds(g0, g1, z4, z4, z8, 0);
        __builtin_amdgcn_s_wait_tensorcnt(0);
    }
#else
    {
        const uint4* src = (const uint4*)(Qb + (size_t)(b * S_LEN + q0) * D_DIM);
        uint4* dst = (uint4*)sQ;
        for (int i = tid; i < 16 * D_DIM / 8; i += 256) dst[i] = src[i];
    }
#endif
    if (tid < 16) { sm_m[tid] = -1e30f; sm_l[tid] = 0.0f; }
    __syncthreads();

    v8f acc[8] = {};                          // 16 x 128 f32 output chunk
    const int nkb = (q0 + 16 + 127) >> 7;     // causal: key blocks needed

    for (int kb = 0; kb < nkb * 128; kb += 128) {
        // ---- scores: S = Q * K^T for this wave's 16 keys (K-dim = 1024) ----
        const __bf16* krow = Kb + (size_t)(b * S_LEN + kb + w * 16 + l16) * D_DIM + lhalf * 16;
        const __bf16* qrow = &sQ[l16 * D_DIM + lhalf * 16];
        __builtin_prefetch(krow + 128 * D_DIM, 0, 1);        // next key block

        v16bf a0  = *(const v16bf*)(qrow);
        v16bf bb0 = *(const v16bf*)(krow);
        v8f c = {};
#pragma unroll 4
        for (int ks = 0; ks < 32; ++ks) {
            const int kn = ((ks + 1) & 31) * 32;             // wraps on last iter
            v16bf a1  = *(const v16bf*)(qrow + kn);
            v16bf bb1 = *(const v16bf*)(krow + kn);
            c = wmma_bf16(a0, bb0, c);
            a0 = a1; bb0 = bb1;
        }

        // ---- scale + causal mask ----
        float sv[8];
        const int j = kb + w * 16 + l16;      // key index (per lane)
#pragma unroll
        for (int r = 0; r < 8; ++r) {
            float s = c[r] * scale;
            int q = q0 + r + 8 * lhalf;
            sv[r] = (j > q) ? -1e30f : s;
        }

        // ---- per-wave row max (reduce across 16 key lanes) ----
        float rmax[8];
#pragma unroll
        for (int r = 0; r < 8; ++r) {
            float v = sv[r];
            for (int off = 8; off >= 1; off >>= 1)
                v = fmaxf(v, __shfl_xor(v, off, 32));
            rmax[r] = v;
        }
        if (l16 == 0)
#pragma unroll
            for (int r = 0; r < 8; ++r) sm_pmax[w * 16 + r + 8 * lhalf] = rmax[r];
        __syncthreads();

        // ---- combine maxima across waves, update running max ----
        if (tid < 16) {
            float mb = -1e30f;
#pragma unroll
            for (int w8 = 0; w8 < 8; ++w8) mb = fmaxf(mb, sm_pmax[w8 * 16 + tid]);
            float mold = sm_m[tid];
            float mnew = fmaxf(mold, mb);
            sm_alpha[tid] = __expf(mold - mnew);
            sm_m[tid] = mnew;
        }
        __syncthreads();

        // ---- P = exp(S - m), row sums, stash P tile to LDS as bf16 ----
#pragma unroll
        for (int r = 0; r < 8; ++r) {
            int row = r + 8 * lhalf;
            float p = __expf(sv[r] - sm_m[row]);
            sP[row * 128 + w * 16 + l16] = (__bf16)p;
            float v = p;
            for (int off = 8; off >= 1; off >>= 1)
                v += __shfl_xor(v, off, 32);
            if (l16 == 0) sm_psum[w * 16 + row] = v;
        }
        __syncthreads();

        if (tid < 16) {
            float s = 0.0f;
#pragma unroll
            for (int w8 = 0; w8 < 8; ++w8) s += sm_psum[w8 * 16 + tid];
            sm_l[tid] = sm_l[tid] * sm_alpha[tid] + s;
        }

        // ---- rescale accumulators by alpha ----
        float ar[8];
#pragma unroll
        for (int r = 0; r < 8; ++r) ar[r] = sm_alpha[r + 8 * lhalf];
#pragma unroll
        for (int t = 0; t < 8; ++t)
#pragma unroll
            for (int r = 0; r < 8; ++r) acc[t][r] *= ar[r];

        // ---- O += P (16x128) @ V(kb block, wave's 128 feature cols) ----
        // flattened 32-step pipeline: idx = t*4 + ks
        const __bf16* vbase = Vt + (size_t)b * D_DIM * S_LEN + kb + lhalf * 16;
        v16bf pa0 = *(const v16bf*)&sP[l16 * 128 + lhalf * 16];
        v16bf vb0 = *(const v16bf*)(vbase + (size_t)(w * 128 + l16) * S_LEN);
#pragma unroll
        for (int idx = 0; idx < 32; ++idx) {
            const int nidx = (idx + 1) & 31;                 // wraps on last iter
            const int tn = nidx >> 2, ksn = nidx & 3;
            v16bf pa1 = *(const v16bf*)&sP[l16 * 128 + ksn * 32 + lhalf * 16];
            v16bf vb1 = *(const v16bf*)(vbase + (size_t)(w * 128 + tn * 16 + l16) * S_LEN + ksn * 32);
            acc[idx >> 2] = wmma_bf16(pa0, vb0, acc[idx >> 2]);
            pa0 = pa1; vb0 = vb1;
        }
        __syncthreads();   // sP / sm_* reused next iteration
    }

    // ---- normalize and write output (f32) ----
    float linv[8];
#pragma unroll
    for (int r = 0; r < 8; ++r) linv[r] = 1.0f / sm_l[r + 8 * lhalf];
#pragma unroll
    for (int t = 0; t < 8; ++t)
#pragma unroll
        for (int r = 0; r < 8; ++r) {
            int q = q0 + r + 8 * lhalf;
            out[(size_t)(b * S_LEN + q) * D_DIM + w * 128 + t * 16 + l16] = acc[t][r] * linv[r];
        }
}

// ---------------------------------------------------------------------------
// Launch
// ---------------------------------------------------------------------------
extern "C" void kernel_launch(void* const* d_in, const int* in_sizes, int n_in,
                              void* d_out, int out_size, void* d_ws, size_t ws_size,
                              hipStream_t stream) {
    const float* x  = (const float*)d_in[0];
    const float* Wq = (const float*)d_in[1];
    const float* Wk = (const float*)d_in[2];
    const float* Wv = (const float*)d_in[3];
    float* out = (float*)d_out;

    // workspace layout (bytes)
    char* ws = (char*)d_ws;
    const size_t sz_xb = (size_t)NTOK * D_DIM * 2;          // 16 MB
    const size_t sz_w  = (size_t)3 * D_DIM * D_DIM * 2;     //  6 MB
    const size_t sz_m  = (size_t)NTOK * D_DIM * 2;          // 16 MB each
    __bf16* xb = (__bf16*)(ws);
    __bf16* wt = (__bf16*)(ws + sz_xb);
    __bf16* Qb = (__bf16*)(ws + sz_xb + sz_w);
    __bf16* Kb = (__bf16*)(ws + sz_xb + sz_w + sz_m);
    __bf16* Vt = (__bf16*)(ws + sz_xb + sz_w + 2 * sz_m);

    // 1) convert x to bf16
    {
        int n = NTOK * D_DIM;
        cvt_x_kernel<<<(n + 255) / 256, 256, 0, stream>>>(x, xb, n);
    }
    // 2) convert + transpose weights to bf16
    {
        int n = 3 * D_DIM * D_DIM;
        cvt_w_kernel<<<(n + 255) / 256, 256, 0, stream>>>(Wq, Wk, Wv, wt);
    }
    // 3) QKV projections (WMMA bf16)
    {
        dim3 grid(NTOK / 32, D_DIM / 256, 3);
        gemm_qkv_kernel<<<grid, 256, 0, stream>>>(xb, wt, Qb, Kb, Vt);
    }
    // 4) causal flash attention (WMMA bf16, online softmax)
    {
        dim3 grid(S_LEN / 16, NBATCH);
        flash_attn_kernel<<<grid, 256, 0, stream>>>(Qb, Kb, Vt, out);
    }
}
